// Qwen2MoeDecoderLayer_32942399161074
// MI455X (gfx1250) — matrix operand order repaired and checked by reading, not compiled, not run on hardware
//
#include <hip/hip_runtime.h>

typedef unsigned short u16;
typedef __attribute__((ext_vector_type(16))) __bf16 v16bf;
typedef __attribute__((ext_vector_type(8)))  float  v8f;

constexpr int kS   = 1024;
constexpr int kD   = 2048;
constexpr int kH   = 16;
constexpr int kHD  = 128;
constexpr int kHHD = 2048;   // H*HD
constexpr int kE   = 8;
constexpr int kTOPK= 4;
constexpr int kFI  = 1408;
constexpr int kFS  = 5632;
constexpr int kWIN = 512;

constexpr int BM = 64, BN = 128, BK = 32;
constexpr int LDP = 40;      // padded LDS row (halves): 80B rows -> conflict-free, 16B aligned

union FragU { unsigned u[8]; v16bf v; };

__device__ __forceinline__ u16 f2bf(float x) {
  unsigned u = __float_as_uint(x);
  u += 0x7FFFu + ((u >> 16) & 1u);         // round-to-nearest-even
  return (u16)(u >> 16);
}
__device__ __forceinline__ float bf2f(u16 h) {
  return __uint_as_float(((unsigned)h) << 16);
}
__device__ __forceinline__ v8f vzero() {
  v8f z = {0.f,0.f,0.f,0.f,0.f,0.f,0.f,0.f};
  return z;
}
__device__ __forceinline__ v8f wmma_bf16(v16bf a, v16bf b, v8f c) {
  return __builtin_amdgcn_wmma_f32_16x16x32_bf16(false, a, false, b, (short)0, c, false, false);
}

// ---------------------------------------------------------------------------
// fp32 [R][C] -> bf16 [C][R] transpose-convert (weights, one-time).
// blockIdx.z selects matrix (for per-expert weight stacks).
// ---------------------------------------------------------------------------
__global__ __launch_bounds__(256) void cvt_t(const float* __restrict__ in,
                                             u16* __restrict__ out, int R, int C) {
  __shared__ float tile[32][33];
  const float* inp = in + (size_t)blockIdx.z * R * C;
  u16* outp = out + (size_t)blockIdx.z * R * C;
  int rb = blockIdx.y * 32, cb = blockIdx.x * 32;
  int tx = threadIdx.x, ty = threadIdx.y;            // (32, 8)
  for (int i = ty; i < 32; i += 8)
    tile[i][tx] = inp[(size_t)(rb + i) * C + cb + tx];
  __syncthreads();
  for (int i = ty; i < 32; i += 8)
    outp[(size_t)(cb + i) * R + rb + tx] = f2bf(tile[tx][i]);
}

// ---------------------------------------------------------------------------
// RMSNorm: one 256-thread block per row, bf16 output
// ---------------------------------------------------------------------------
__global__ __launch_bounds__(256) void rmsnorm_bf16(const float* __restrict__ x,
                                                    const float* __restrict__ w,
                                                    u16* __restrict__ out) {
  __shared__ float red[256];
  const int row = blockIdx.x;
  const float* xr = x + (size_t)row * kD;
  float ss = 0.f;
  for (int c = threadIdx.x; c < kD; c += 256) { float v = xr[c]; ss += v * v; }
  red[threadIdx.x] = ss;
  __syncthreads();
  for (int off = 128; off > 0; off >>= 1) {
    if (threadIdx.x < off) red[threadIdx.x] += red[threadIdx.x + off];
    __syncthreads();
  }
  float inv = rsqrtf(red[0] / (float)kD + 1e-6f);
  for (int c = threadIdx.x; c < kD; c += 256)
    out[(size_t)row * kD + c] = f2bf(xr[c] * inv * w[c]);
}

// ---------------------------------------------------------------------------
// bf16 WMMA GEMM: C[M,N] = A[M,K] @ B[K,N] (+bias), B pre-transposed as Bt[N][K].
// 128 threads = 4 wave32; tile 64x128, K-step 32; per-wave 16x128 strip.
// Tiles staged with global_load_async_to_lds_b128, double-buffered via ASYNCcnt.
// ---------------------------------------------------------------------------
__global__ __launch_bounds__(128) void gemm_bf16(const u16* __restrict__ A,
                                                 const u16* __restrict__ Bt,
                                                 const float* __restrict__ bias,
                                                 float* __restrict__ C,
                                                 int N, int K) {
  __shared__ u16 As[2][BM][LDP];
  __shared__ u16 Bs[2][BN][LDP];
  const int tid = threadIdx.x;
  const int lane = tid & 31, wave = tid >> 5;
  const int hi = (lane >> 4) & 1, l16 = lane & 15;
  const int mBase = blockIdx.y * BM;
  const int nBase = blockIdx.x * BN;

  // per-thread 16B-chunk coordinates for cooperative tile loads
  unsigned a_voff[2], b_voff[4];
#pragma unroll
  for (int it = 0; it < 2; ++it) {
    int vi = tid + it * 128, r = vi >> 2, c = (vi & 3) * 8;
    a_voff[it] = (unsigned)((((size_t)(mBase + r)) * K + c) * 2);
  }
#pragma unroll
  for (int it = 0; it < 4; ++it) {
    int vi = tid + it * 128, r = vi >> 2, c = (vi & 3) * 8;
    b_voff[it] = (unsigned)((((size_t)(nBase + r)) * K + c) * 2);
  }

  auto issue = [&](int buf, int k0) {
    unsigned long long abase = (unsigned long long)(size_t)(A + k0);
    unsigned long long bbase = (unsigned long long)(size_t)(Bt + k0);
#pragma unroll
    for (int it = 0; it < 2; ++it) {
      int vi = tid + it * 128, r = vi >> 2, c = (vi & 3) * 8;
      unsigned lds = (unsigned)(size_t)&As[buf][r][c];   // flat addr low32 == LDS offset
      asm volatile("global_load_async_to_lds_b128 %0, %1, %2"
                   :: "v"(lds), "v"(a_voff[it]), "s"(abase) : "memory");
    }
#pragma unroll
    for (int it = 0; it < 4; ++it) {
      int vi = tid + it * 128, r = vi >> 2, c = (vi & 3) * 8;
      unsigned lds = (unsigned)(size_t)&Bs[buf][r][c];
      asm volatile("global_load_async_to_lds_b128 %0, %1, %2"
                   :: "v"(lds), "v"(b_voff[it]), "s"(bbase) : "memory");
    }
  };

  v8f acc[8];
#pragma unroll
  for (int j = 0; j < 8; ++j) acc[j] = vzero();

  const int nt = K / BK;
  issue(0, 0);
  for (int i = 0; i < nt; ++i) {
    const int cur = i & 1;
    if (i + 1 < nt) {
      issue(cur ^ 1, (i + 1) * BK);                     // prefetch next tile
      asm volatile("s_wait_asynccnt 6" ::: "memory");   // wait only for current tile's 6
    } else {
      asm volatile("s_wait_asynccnt 0" ::: "memory");
    }
    __syncthreads();

    FragU af;                                           // A frag 16x32: 2x b128 per lane
#pragma unroll
    for (int v = 0; v < 8; ++v) {
      int kp = 2 * v + hi * 16;
      af.u[v] = *(const unsigned*)&As[cur][wave * 16 + l16][kp];
    }
#pragma unroll
    for (int j = 0; j < 8; ++j) {                       // B frags: 2x b128 per lane each
      FragU bfr;
#pragma unroll
      for (int w = 0; w < 8; ++w)
        bfr.u[w] = *(const unsigned*)&Bs[cur][j * 16 + l16][2 * w + hi * 16];
      acc[j] = wmma_bf16(af.v, bfr.v, acc[j]);
    }
    __syncthreads();
  }

#pragma unroll
  for (int j = 0; j < 8; ++j)
#pragma unroll
    for (int r = 0; r < 8; ++r) {
      int row = mBase + wave * 16 + r + hi * 8;
      int col = nBase + j * 16 + l16;
      float v = acc[j][r];
      if (bias) v += bias[col];
      C[(size_t)row * N + col] = v;
    }
}

// ---------------------------------------------------------------------------
// RoPE + bf16 conversion; V additionally written TRANSPOSED: vt[(head*HD+hd)][s]
// ---------------------------------------------------------------------------
__global__ __launch_bounds__(256) void rope_cvt(const float* __restrict__ qf,
                                                const float* __restrict__ kf,
                                                const float* __restrict__ vf,
                                                u16* __restrict__ qb,
                                                u16* __restrict__ kb,
                                                u16* __restrict__ vtb) {
  int idx = blockIdx.x * 256 + threadIdx.x;     // over S*HHD (exact grid)
  int srow = idx / kHHD;
  int c    = idx & (kHHD - 1);
  int hd   = c & (kHD - 1);
  int dd   = hd & 63;
  float inv = __expf(-13.815510557964274f * (float)dd * (1.0f / 64.0f));
  float ang = (float)srow * inv;
  float sn, cs;
  __sincosf(ang, &sn, &cs);
  float qv = qf[idx], kv = kf[idx], qo, ko;
  if (hd < 64) { qo = qv * cs - qf[idx + 64] * sn; ko = kv * cs - kf[idx + 64] * sn; }
  else         { qo = qv * cs + qf[idx - 64] * sn; ko = kv * cs + kf[idx - 64] * sn; }
  qb[idx] = f2bf(qo); kb[idx] = f2bf(ko);
  vtb[(size_t)c * kS + srow] = f2bf(vf[idx]);
}

// ---------------------------------------------------------------------------
// Sliding-window causal attention; QK^T and P.V on WMMA.
// One wave32 per (16-query tile, head). V read from transposed layout.
// ---------------------------------------------------------------------------
__global__ __launch_bounds__(32) void attn_swa(const u16* __restrict__ Q,
                                               const u16* __restrict__ Kb,
                                               const u16* __restrict__ Vt,
                                               u16* __restrict__ O) {
  __shared__ float sc[16][544];
  __shared__ u16   Ps[16][544];
  __shared__ float rsum[16];

  const int lane = threadIdx.x & 31;
  const int hi = (lane >> 4) & 1, l16 = lane & 15;
  const int q0 = blockIdx.x * 16;
  const int head = blockIdx.y;
  const float scale = 0.08838834764831845f;     // 1/sqrt(128)

  int jb0 = q0 - 512; if (jb0 < 0) jb0 = 0;
  const int L   = q0 + 16 - jb0;                // mult of 16, <=528
  const int nb  = L / 16;
  const int Lp  = ((L + 31) >> 5) << 5;
  const int nbp = Lp >> 5;

  FragU qfr[4];
#pragma unroll
  for (int s4 = 0; s4 < 4; ++s4)
#pragma unroll
    for (int v = 0; v < 8; ++v) {
      int kp = 2 * v + hi * 16;
      qfr[s4].u[v] = *(const unsigned*)(Q + (size_t)(q0 + l16) * kHHD + head * kHD + s4 * 32 + kp);
    }

  for (int b = 0; b < nb; ++b) {
    const int jb = jb0 + b * 16;
    v8f s = vzero();
#pragma unroll
    for (int s4 = 0; s4 < 4; ++s4) {
      FragU kfr;                                 // B frag: N=key(lane), K=hd (contiguous)
#pragma unroll
      for (int w = 0; w < 8; ++w) {
        int khd = 2 * w + hi * 16;
        kfr.u[w] = *(const unsigned*)(Kb + (size_t)(jb + l16) * kHHD + head * kHD + s4 * 32 + khd);
      }
      s = wmma_bf16(qfr[s4].v, kfr.v, s);
    }
#pragma unroll
    for (int r = 0; r < 8; ++r) {
      int il = r + hi * 8;
      int i = q0 + il;
      int j = jb + l16;
      float val = s[r] * scale;
      if (j > i || (i - j) >= kWIN) val = -3.0e38f;
      sc[il][b * 16 + l16] = val;
    }
  }
  __syncthreads();

  if (lane < 16) {
    float m = -3.4e38f;
    for (int c = 0; c < L; ++c) m = fmaxf(m, sc[lane][c]);
    float sum = 0.f;
    for (int c = 0; c < L; ++c) {
      float ev = __expf(sc[lane][c] - m);
      sum += ev;
      Ps[lane][c] = f2bf(ev);
    }
    for (int c = L; c < Lp; ++c) Ps[lane][c] = 0;
    rsum[lane] = sum;
  }
  __syncthreads();

  v8f o[8];
#pragma unroll
  for (int n = 0; n < 8; ++n) o[n] = vzero();

  for (int b2 = 0; b2 < nbp; ++b2) {
    FragU pf;                                    // A frag from P (16x32), b32 LDS reads
#pragma unroll
    for (int v = 0; v < 8; ++v) {
      int kp = 2 * v + hi * 16;
      pf.u[v] = *(const unsigned*)&Ps[l16][b2 * 32 + kp];
    }
#pragma unroll
    for (int n = 0; n < 8; ++n) {                // B frags from transposed V: key-contiguous
      FragU vfr;
#pragma unroll
      for (int w = 0; w < 8; ++w) {
        int key = jb0 + b2 * 32 + 2 * w + hi * 16;
        if (key > kS - 2) key = kS - 2;          // OOB pad: P==0 there, data finite
        vfr.u[w] = *(const unsigned*)(Vt + (size_t)(head * kHD + n * 16 + l16) * kS + key);
      }
      o[n] = wmma_bf16(pf.v, vfr.v, o[n]);
    }
  }

#pragma unroll
  for (int n = 0; n < 8; ++n)
#pragma unroll
    for (int r = 0; r < 8; ++r) {
      int il = r + hi * 8;
      float val = o[n][r] / rsum[il];
      O[(size_t)(q0 + il) * kHHD + head * kHD + n * 16 + l16] = f2bf(val);
    }
}

// ---------------------------------------------------------------------------
// Router: logits -> softmax -> top-4 combine weights (norm_topk_prob=False)
// ---------------------------------------------------------------------------
__global__ __launch_bounds__(128) void router_topk(const u16* __restrict__ h,
                                                   const float* __restrict__ wr,
                                                   float* __restrict__ cw) {
  int t = blockIdx.x * 128 + threadIdx.x;
  if (t >= kS) return;
  float lg[kE];
#pragma unroll
  for (int e = 0; e < kE; ++e) lg[e] = 0.f;
  const u16* hr = h + (size_t)t * kD;
  for (int d = 0; d < kD; ++d) {
    float hv = bf2f(hr[d]);
#pragma unroll
    for (int e = 0; e < kE; ++e) lg[e] += hv * wr[d * kE + e];
  }
  float m = lg[0];
  for (int e = 1; e < kE; ++e) m = fmaxf(m, lg[e]);
  float p[kE], sum = 0.f;
  for (int e = 0; e < kE; ++e) { p[e] = __expf(lg[e] - m); sum += p[e]; }
  for (int e = 0; e < kE; ++e) p[e] /= sum;
  float outw[kE];
  for (int e = 0; e < kE; ++e) outw[e] = 0.f;
  unsigned used = 0;
  for (int k = 0; k < kTOPK; ++k) {
    int bi = 0; float bv = -1.f;
    for (int e = 0; e < kE; ++e)
      if (!((used >> e) & 1u) && p[e] > bv) { bv = p[e]; bi = e; }
    used |= 1u << bi; outw[bi] = bv;
  }
  for (int e = 0; e < kE; ++e) cw[(size_t)t * kE + e] = outw[e];
}

// ---------------------------------------------------------------------------
// elementwise helpers
// ---------------------------------------------------------------------------
__global__ __launch_bounds__(256) void silu_mul(const float* __restrict__ g,
                                                const float* __restrict__ u,
                                                u16* __restrict__ y, int n) {
  int i = blockIdx.x * 256 + threadIdx.x;
  if (i >= n) return;
  float gv = g[i];
  float s = gv / (1.f + __expf(-gv));
  y[i] = f2bf(s * u[i]);
}

__global__ __launch_bounds__(256) void moe_combine(const float* __restrict__ eo,
                                                   const float* __restrict__ cw,
                                                   float* __restrict__ routed,
                                                   int e, int init) {
  int i = blockIdx.x * 256 + threadIdx.x;
  if (i >= kS * kD) return;
  int t = i / kD;
  float v = cw[(size_t)t * kE + e] * eo[i];
  routed[i] = init ? v : (routed[i] + v);
}

__global__ __launch_bounds__(256) void add2(const float* __restrict__ a,
                                            const float* __restrict__ b,
                                            float* __restrict__ c, int n) {
  int i = blockIdx.x * 256 + threadIdx.x;
  if (i < n) c[i] = a[i] + b[i];
}

__global__ __launch_bounds__(128) void shared_gate(const u16* __restrict__ h,
                                                   const float* __restrict__ wg,
                                                   float* __restrict__ sg) {
  int t = blockIdx.x * 128 + threadIdx.x;
  if (t >= kS) return;
  const u16* hr = h + (size_t)t * kD;
  float acc = 0.f;
  for (int d = 0; d < kD; ++d) acc += bf2f(hr[d]) * wg[d];
  sg[t] = 1.f / (1.f + __expf(-acc));
}

__global__ __launch_bounds__(256) void final_out(const float* __restrict__ x1,
                                                 const float* __restrict__ routed,
                                                 const float* __restrict__ sh,
                                                 const float* __restrict__ sg,
                                                 float* __restrict__ out) {
  int i = blockIdx.x * 256 + threadIdx.x;
  if (i >= kS * kD) return;
  int t = i / kD;
  out[i] = x1[i] + routed[i] + sg[t] * sh[i];
}

// ---------------------------------------------------------------------------
// host orchestration
// ---------------------------------------------------------------------------
extern "C" void kernel_launch(void* const* d_in, const int* in_sizes, int n_in,
                              void* d_out, int out_size, void* d_ws, size_t ws_size,
                              hipStream_t stream) {
  (void)in_sizes; (void)n_in; (void)out_size; (void)ws_size;

  const float* x       = (const float*)d_in[0];
  const float* ln1w    = (const float*)d_in[1];
  const float* ln2w    = (const float*)d_in[2];
  const float* wq      = (const float*)d_in[3];
  const float* bq      = (const float*)d_in[4];
  const float* wk      = (const float*)d_in[5];
  const float* bk      = (const float*)d_in[6];
  const float* wv      = (const float*)d_in[7];
  const float* bv      = (const float*)d_in[8];
  const float* wo      = (const float*)d_in[9];
  const float* wrouter = (const float*)d_in[10];
  const float* wge     = (const float*)d_in[11];
  const float* wue     = (const float*)d_in[12];
  const float* wde     = (const float*)d_in[13];
  const float* wsg     = (const float*)d_in[14];
  const float* wsu     = (const float*)d_in[15];
  const float* wsd     = (const float*)d_in[16];
  const float* wshg    = (const float*)d_in[17];
  float* out = (float*)d_out;

  char* base = (char*)d_ws;
  size_t off = 0;
  auto alloc = [&](size_t elems, size_t esz) -> void* {
    off = (off + 255) & ~(size_t)255;
    void* p = base + off;
    off += elems * esz;
    return p;
  };

  // bf16 scratch (all weight matrices stored TRANSPOSED: [N][K])
  u16* wq_b  = (u16*)alloc((size_t)kD * kHHD, 2);
  u16* wk_b  = (u16*)alloc((size_t)kD * kHHD, 2);
  u16* wv_b  = (u16*)alloc((size_t)kD * kHHD, 2);
  u16* wo_b  = (u16*)alloc((size_t)kHHD * kD, 2);
  u16* wge_b = (u16*)alloc((size_t)kE * kD * kFI, 2);
  u16* wue_b = (u16*)alloc((size_t)kE * kD * kFI, 2);
  u16* wde_b = (u16*)alloc((size_t)kE * kFI * kD, 2);
  u16* wsg_b = (u16*)alloc((size_t)kD * kFS, 2);
  u16* wsu_b = (u16*)alloc((size_t)kD * kFS, 2);
  u16* wsd_b = (u16*)alloc((size_t)kFS * kD, 2);
  u16* h1_b  = (u16*)alloc((size_t)kS * kD, 2);
  u16* h2_b  = (u16*)alloc((size_t)kS * kD, 2);
  u16* qb    = (u16*)alloc((size_t)kS * kHHD, 2);
  u16* kbuf  = (u16*)alloc((size_t)kS * kHHD, 2);
  u16* vtb   = (u16*)alloc((size_t)kHHD * kS, 2);   // transposed V
  u16* attnb = (u16*)alloc((size_t)kS * kHHD, 2);
  u16* y_b   = (u16*)alloc((size_t)kS * kFI, 2);
  u16* ys_b  = (u16*)alloc((size_t)kS * kFS, 2);
  // fp32 scratch
  float* qfo    = (float*)alloc((size_t)kS * kHHD, 4);
  float* kfo    = (float*)alloc((size_t)kS * kHHD, 4);
  float* vfo    = (float*)alloc((size_t)kS * kHHD, 4);
  float* ofp    = (float*)alloc((size_t)kS * kD, 4);
  float* x1     = (float*)alloc((size_t)kS * kD, 4);
  float* gbuf   = (float*)alloc((size_t)kS * kFI, 4);
  float* ubuf   = (float*)alloc((size_t)kS * kFI, 4);
  float* eobuf  = (float*)alloc((size_t)kS * kD, 4);
  float* routed = (float*)alloc((size_t)kS * kD, 4);
  float* gsf    = (float*)alloc((size_t)kS * kFS, 4);
  float* usf    = (float*)alloc((size_t)kS * kFS, 4);
  float* shf    = (float*)alloc((size_t)kS * kD, 4);
  float* cw     = (float*)alloc((size_t)kS * kE, 4);
  float* sg     = (float*)alloc((size_t)kS, 4);

  auto cvtT = [&](const float* in, u16* o, int R, int C, int Z) {
    cvt_t<<<dim3(C / 32, R / 32, Z), dim3(32, 8), 0, stream>>>(in, o, R, C);
  };
  auto gemm = [&](const u16* A, const u16* Bt, const float* bias, float* C,
                  int M, int N, int K) {
    dim3 grid(N / BN, M / BM);
    gemm_bf16<<<grid, 128, 0, stream>>>(A, Bt, bias, C, N, K);
  };

  // one-time weight transpose-convert fp32 [K][N] -> bf16 [N][K]
  cvtT(wq, wq_b, kD, kHHD, 1);
  cvtT(wk, wk_b, kD, kHHD, 1);
  cvtT(wv, wv_b, kD, kHHD, 1);
  cvtT(wo, wo_b, kHHD, kD, 1);
  cvtT(wge, wge_b, kD, kFI, kE);
  cvtT(wue, wue_b, kD, kFI, kE);
  cvtT(wde, wde_b, kFI, kD, kE);
  cvtT(wsg, wsg_b, kD, kFS, 1);
  cvtT(wsu, wsu_b, kD, kFS, 1);
  cvtT(wsd, wsd_b, kFS, kD, 1);

  // --- attention block ---
  rmsnorm_bf16<<<kS, 256, 0, stream>>>(x, ln1w, h1_b);
  gemm(h1_b, wq_b, bq, qfo, kS, kHHD, kD);
  gemm(h1_b, wk_b, bk, kfo, kS, kHHD, kD);
  gemm(h1_b, wv_b, bv, vfo, kS, kHHD, kD);
  rope_cvt<<<(kS * kHHD) / 256, 256, 0, stream>>>(qfo, kfo, vfo, qb, kbuf, vtb);
  attn_swa<<<dim3(kS / 16, kH), 32, 0, stream>>>(qb, kbuf, vtb, attnb);
  gemm(attnb, wo_b, nullptr, ofp, kS, kD, kHHD);
  add2<<<(kS * kD) / 256, 256, 0, stream>>>(x, ofp, x1, kS * kD);

  // --- MoE block ---
  rmsnorm_bf16<<<kS, 256, 0, stream>>>(x1, ln2w, h2_b);
  router_topk<<<kS / 128, 128, 0, stream>>>(h2_b, wrouter, cw);
  for (int e = 0; e < kE; ++e) {  // dense combine (cw==0 for unselected) == reference
    gemm(h2_b, wge_b + (size_t)e * kD * kFI, nullptr, gbuf, kS, kFI, kD);
    gemm(h2_b, wue_b + (size_t)e * kD * kFI, nullptr, ubuf, kS, kFI, kD);
    silu_mul<<<(kS * kFI) / 256, 256, 0, stream>>>(gbuf, ubuf, y_b, kS * kFI);
    gemm(y_b, wde_b + (size_t)e * kFI * kD, nullptr, eobuf, kS, kD, kFI);
    moe_combine<<<(kS * kD) / 256, 256, 0, stream>>>(eobuf, cw, routed, e, e == 0 ? 1 : 0);
  }
  // shared expert
  gemm(h2_b, wsg_b, nullptr, gsf, kS, kFS, kD);
  gemm(h2_b, wsu_b, nullptr, usf, kS, kFS, kD);
  silu_mul<<<(kS * kFS) / 256, 256, 0, stream>>>(gsf, usf, ys_b, kS * kFS);
  gemm(ys_b, wsd_b, nullptr, shf, kS, kD, kFS);
  shared_gate<<<kS / 128, 128, 0, stream>>>(h2_b, wshg, sg);

  final_out<<<(kS * kD) / 256, 256, 0, stream>>>(x1, routed, shf, sg, out);
}
